// MixtureOfExperts_24541443130012
// MI455X (gfx1250) — compile-verified, matrix operand order checked by reading, not run on hardware
//
#include <hip/hip_runtime.h>
#include <hip/hip_bf16.h>

typedef __bf16 v16bf __attribute__((ext_vector_type(16)));
typedef __bf16 v8bf  __attribute__((ext_vector_type(8)));
typedef float  v8f   __attribute__((ext_vector_type(8)));

static constexpr int Bq = 4, Sq = 2048, Dq = 512, Fq = 2048, Eq = 8;
static constexpr int T   = Bq * Sq;   // 8192 tokens
static constexpr int MT  = 64;        // tokens per workgroup (4 WMMA row tiles)
static constexpr int FCH = 256;       // F chunk staged in LDS
static constexpr int XP  = Dq + 8;    // padded X row (bf16 elems), 16B-aligned stride
static constexpr int HP  = FCH + 8;   // padded H row
static constexpr int SMEM_BYTES = MT * XP * 2 + MT * HP * 2 + MT * 4;

// ---- bf16 fragment load: two contiguous 16B loads per lane ----
__device__ __forceinline__ v16bf ldfrag(const __bf16* p) {
    union { v16bf v; v8bf h[2]; } u;
    u.h[0] = *(const v8bf*)(p);
    u.h[1] = *(const v8bf*)(p + 16);
    return u.v;
}

// ---- zero-init output ----
__global__ void k_zero(float* __restrict__ p, int n) {
    int i = blockIdx.x * 256 + threadIdx.x;
    if (i < n) p[i] = 0.f;
}

// ---- convert x to bf16 ----
__global__ void k_cvt(const float* __restrict__ src, __bf16* __restrict__ dst, int n) {
    int i = blockIdx.x * 256 + threadIdx.x;
    if (i < n) dst[i] = (__bf16)src[i];
}

// ---- transpose+convert: src [E][R][C] f32 -> dst [E][C][R] bf16 (coalesced writes) ----
__global__ void k_tr(const float* __restrict__ src, __bf16* __restrict__ dst, int R, int C) {
    size_t i = (size_t)blockIdx.x * 256 + threadIdx.x;      // over E*R*C
    size_t RC = (size_t)R * C;
    size_t e = i / RC;
    size_t rem = i - e * RC;
    int r = (int)(rem % R);
    int c = (int)(rem / R);
    dst[i] = (__bf16)src[e * RC + (size_t)r * C + c];
}

// ---- gating: logits, top-2, softmax -> dense gw[t][8] ----
__global__ void k_gate(const float* __restrict__ x, const float* __restrict__ Wg,
                       float* __restrict__ gw) {
    int t = blockIdx.x * 256 + threadIdx.x;
    if (t >= T) return;
    float acc[Eq];
#pragma unroll
    for (int e = 0; e < Eq; ++e) acc[e] = 0.f;
    const float* xt = x + (size_t)t * Dq;
    for (int d = 0; d < Dq; ++d) {
        float xv = xt[d];
        const float* wr = Wg + d * Eq;
#pragma unroll
        for (int e = 0; e < Eq; ++e) acc[e] = fmaf(xv, wr[e], acc[e]);
    }
    int i1 = 0; float v1 = acc[0];
#pragma unroll
    for (int e = 1; e < Eq; ++e) if (acc[e] > v1) { v1 = acc[e]; i1 = e; }
    int i2 = 0; float v2 = -3.0e38f;
#pragma unroll
    for (int e = 0; e < Eq; ++e) if (e != i1 && acc[e] > v2) { v2 = acc[e]; i2 = e; }
    float ex = __expf(v2 - v1);
    float inv = 1.f / (1.f + ex);
    float* g = gw + (size_t)t * Eq;
#pragma unroll
    for (int e = 0; e < Eq; ++e) g[e] = 0.f;
    g[i1] = inv;
    g[i2] = ex * inv;
}

// ---- MoE main: 64 tokens x 1 expert per workgroup; atomic combine into y ----
__global__ __launch_bounds__(256) void k_moe(
        const __bf16* __restrict__ xb,     // [T][D]
        const __bf16* __restrict__ Wt,     // [E][F][D]  (W transposed)
        const __bf16* __restrict__ Vt,     // [E][F][D]
        const __bf16* __restrict__ W2t,    // [E][D][F]  (W2 transposed)
        const float*  __restrict__ gw,     // [T][E]
        float* __restrict__ y) {           // [T][D], pre-zeroed
    extern __shared__ __align__(16) char smem[];
    __bf16* Xs  = (__bf16*)smem;           // [MT][XP]
    __bf16* Hs  = Xs + MT * XP;            // [MT][HP]
    float*  gws = (float*)(Hs + MT * HP);  // [MT]

    const int tid   = threadIdx.x;
    const int wv    = tid >> 5;            // wave 0..7
    const int lane  = tid & 31;
    const int lidx  = lane & 15;
    const int k0    = (lane >> 4) * 8;     // K sub-slice select
    const int m0    = (lane >> 4) * 8;     // C/D row base
    const int e     = blockIdx.x & 7;
    const int tbase = (blockIdx.x >> 3) * MT;

    // ---- stage X tile (64x512 bf16) into LDS; load gate weights ----
    {
        const int r  = tid >> 2;           // 0..63
        const int cb = (tid & 3) * 128;
        const __bf16* src = xb + (size_t)(tbase + r) * Dq + cb;
        __bf16* dst = Xs + r * XP + cb;
#pragma unroll
        for (int c = 0; c < 128; c += 8)
            *(v8bf*)(dst + c) = *(const v8bf*)(src + c);
    }
    if (tid < MT) gws[tid] = gw[(size_t)(tbase + tid) * Eq + e];

    v8f acc[4][4];                         // [n-tile][m-tile]
#pragma unroll
    for (int n = 0; n < 4; ++n)
#pragma unroll
        for (int m = 0; m < 4; ++m)
            acc[n][m] = (v8f){0.f,0.f,0.f,0.f,0.f,0.f,0.f,0.f};

    for (int ch = 0; ch < Fq / FCH; ++ch) {
        __syncthreads();   // Xs/gws ready; previous phase-2 done with Hs
        // ---- phase 1: H chunk [64][FCH] = silu(X W) * (X V), row-scaled by gw ----
        for (int nt = 0; nt < 2; ++nt) {
            const int fl   = wv * 32 + nt * 16 + lidx;   // local col in chunk
            const int fcol = ch * FCH + fl;
            const __bf16* brW = Wt + ((size_t)e * Fq + fcol) * Dq;
            const __bf16* brV = Vt + ((size_t)e * Fq + fcol) * Dq;
            v8f aw[4], av[4];
#pragma unroll
            for (int m = 0; m < 4; ++m) {
                aw[m] = (v8f){0.f,0.f,0.f,0.f,0.f,0.f,0.f,0.f};
                av[m] = (v8f){0.f,0.f,0.f,0.f,0.f,0.f,0.f,0.f};
            }
            for (int kk = 0; kk < Dq; kk += 32) {
                v16bf Bw = ldfrag(brW + kk + k0);
                v16bf Bv = ldfrag(brV + kk + k0);
#pragma unroll
                for (int mt = 0; mt < 4; ++mt) {
                    v16bf A = ldfrag(Xs + (mt * 16 + lidx) * XP + kk + k0);
                    aw[mt] = __builtin_amdgcn_wmma_f32_16x16x32_bf16(
                                 false, A, false, Bw, (short)0, aw[mt], false, false);
                    av[mt] = __builtin_amdgcn_wmma_f32_16x16x32_bf16(
                                 false, A, false, Bv, (short)0, av[mt], false, false);
                }
            }
#pragma unroll
            for (int mt = 0; mt < 4; ++mt)
#pragma unroll
                for (int i = 0; i < 8; ++i) {
                    float a = aw[mt][i], b = av[mt][i];
                    float h = (a / (1.f + __expf(-a))) * b;      // silu(a)*b
                    Hs[(mt * 16 + m0 + i) * HP + fl] =
                        (__bf16)(h * gws[mt * 16 + m0 + i]);
                }
        }
        __syncthreads();   // Hs ready
        // ---- phase 2: acc[64][64 per wave] += H_chunk @ W2[e] chunk ----
        for (int kk = 0; kk < FCH; kk += 32) {
            v16bf A[4];
#pragma unroll
            for (int mt = 0; mt < 4; ++mt)
                A[mt] = ldfrag(Hs + (mt * 16 + lidx) * HP + kk + k0);
#pragma unroll
            for (int nt = 0; nt < 4; ++nt) {
                const int dcol = wv * 64 + nt * 16 + lidx;
                v16bf Bf = ldfrag(W2t + ((size_t)e * Dq + dcol) * Fq
                                  + ch * FCH + kk + k0);
#pragma unroll
                for (int mt = 0; mt < 4; ++mt)
                    acc[nt][mt] = __builtin_amdgcn_wmma_f32_16x16x32_bf16(
                                      false, A[mt], false, Bf, (short)0,
                                      acc[nt][mt], false, false);
            }
        }
    }
    // ---- combine across experts: atomic f32 adds (processed at L2) ----
#pragma unroll
    for (int nt = 0; nt < 4; ++nt) {
        const int dcol = wv * 64 + nt * 16 + lidx;
#pragma unroll
        for (int mt = 0; mt < 4; ++mt)
#pragma unroll
            for (int i = 0; i < 8; ++i)
                atomicAdd(&y[(size_t)(tbase + mt * 16 + m0 + i) * Dq + dcol],
                          acc[nt][mt][i]);
    }
}

extern "C" void kernel_launch(void* const* d_in, const int* in_sizes, int n_in,
                              void* d_out, int out_size, void* d_ws, size_t ws_size,
                              hipStream_t stream) {
    (void)in_sizes; (void)n_in; (void)out_size; (void)ws_size;
    const float* x  = (const float*)d_in[0];
    const float* Wg = (const float*)d_in[1];
    const float* W  = (const float*)d_in[2];
    const float* V  = (const float*)d_in[3];
    const float* W2 = (const float*)d_in[4];
    float* y = (float*)d_out;

    const size_t nX = (size_t)T * Dq;          // 4,194,304
    const size_t nW = (size_t)Eq * Dq * Fq;    // 8,388,608
    __bf16* xb  = (__bf16*)d_ws;
    __bf16* Wt  = xb  + nX;
    __bf16* Vt  = Wt  + nW;
    __bf16* W2t = Vt  + nW;
    float*  gwp = (float*)(W2t + nW);

    k_zero<<<(int)(nX / 256), 256, 0, stream>>>(y, (int)nX);
    k_cvt<<<(int)(nX / 256), 256, 0, stream>>>(x, xb, (int)nX);
    k_tr <<<(int)(nW / 256), 256, 0, stream>>>(W,  Wt,  Dq, Fq);   // [E][D][F]->[E][F][D]
    k_tr <<<(int)(nW / 256), 256, 0, stream>>>(V,  Vt,  Dq, Fq);
    k_tr <<<(int)(nW / 256), 256, 0, stream>>>(W2, W2t, Fq, Dq);   // [E][F][D]->[E][D][F]
    k_gate<<<T / 256, 256, 0, stream>>>(x, Wg, gwp);
    k_moe<<<(T / MT) * Eq, 256, SMEM_BYTES, stream>>>(xb, Wt, Vt, W2t, gwp, y);
}
